// MambaAudioLMBackbone_90202903150842
// MI455X (gfx1250) — compile-verified, matrix operand order checked by reading, not run on hardware
//
#include <hip/hip_runtime.h>
#include <hip/hip_bf16.h>

// ---------------------------------------------------------------------------
// MambaAudioLMBackbone forward for gfx1250 (MI455X).
//   B=2, S=512, P=64, T=576, D=1024, DI=2048, DS=16, DC=4, DR=64, PCD=768, L=4
// GEMMs run on V_WMMA_F32_16X16X32_BF16 (fp32 inputs converted to bf16 in
// LDS, f32 accumulation). Selective scan is fused (states in registers) to
// avoid materializing the [B,T,DI,DS] dA/dBu tensors (~300MB of HBM traffic).
// LDS tiles are stored fragment-friendly: per-lane WMMA fragments are two
// contiguous 16B chunks -> ds_load_b128 pairs. The GEMM is templated per call
// site so no dead guarded staging code is emitted (M and K are tile-aligned
// for every GEMM in this model; only x_proj has a partial N=96). f32->bf16
// uses the hardware packed convert when the toolchain exposes it.
// ---------------------------------------------------------------------------

typedef __attribute__((ext_vector_type(16))) __bf16 v16bf;
typedef __attribute__((ext_vector_type(8)))  __bf16 v8bf;
typedef __attribute__((ext_vector_type(8)))  float  v8f;

#define TILE_M 64
#define TILE_N 64
#define TILE_K 32
#define LDS_STRIDE 40   // halves; 80B row stride -> 16B aligned K-groups

__device__ inline unsigned pack_bf16x2(float a, float b) {
#if __has_builtin(__builtin_amdgcn_cvt_pk_bf16_f32)
    auto p = __builtin_amdgcn_cvt_pk_bf16_f32(a, b);   // packed hw convert
    return __builtin_bit_cast(unsigned, p);
#else
    __bf16 lo = (__bf16)a, hi = (__bf16)b;             // backend fptrunc
    return (unsigned)__builtin_bit_cast(unsigned short, lo) |
           ((unsigned)__builtin_bit_cast(unsigned short, hi) << 16);
#endif
}
__device__ inline __bf16 f2bf(float f) { return (__bf16)f; }

__device__ inline float silu_f(float v)  { return v / (1.0f + __expf(-v)); }
__device__ inline float softplus_f(float v) {
    return (v > 20.0f) ? v : log1pf(__expf(v));
}

// ---------------------------------------------------------------------------
// WMMA GEMM:  C[M,N] = ACT( A[M,K] * op(W) + bias ) + residual
//   WT == 0 : W stored [N,K] (compute A @ W^T)   -- all per-layer weights
//   WT == 1 : W stored [K,N] (compute A @ W)     -- prepend MLP weights
// Template contract (true for every launch in this model):
//   M % 64 == 0, K % 32 == 0; N % 64 == 0 iff FULL_N.
// Block: 128 threads (4 waves). Block tile 64x64; each wave owns 32x32 as a
// 2x2 grid of 16x16 WMMA tiles.
// ---------------------------------------------------------------------------
template <bool FULL_N, int WT, int ACT, bool BIAS, bool RES>
__global__ __launch_bounds__(128) void gemm_wmma_kernel(
    const float* __restrict__ A, int lda,
    const float* __restrict__ W, int ldw,
    float* __restrict__ C, int ldc,
    const float* __restrict__ bias,
    const float* __restrict__ residual,
    int N, int K)
{
    __shared__ __bf16 As [TILE_M][LDS_STRIDE];   // As[m][k]
    __shared__ __bf16 BsT[TILE_N][LDS_STRIDE];   // BsT[n][k]

    const int tid  = threadIdx.x;
    const int wave = tid >> 5;
    const int lane = tid & 31;
    const int mBase = blockIdx.y * TILE_M;
    const int nBase = blockIdx.x * TILE_N;
    const int wm = (wave >> 1) * 32;
    const int wn = (wave & 1) * 32;

    const int halfSel = lane >> 4;   // selects K-half (A/B) and M-half (C/D)
    const int l16     = lane & 15;
    const int kb      = halfSel * 8;

    // staging thread maps (row-contiguous float4 loads)
    const int sr  = tid >> 3;       // 0..15 : row within 16-row slab
    const int sc  = tid & 7;        // 0..7  : float4 column group
    const int tn4 = tid & 15;       // WT=1 map: n-group of 4
    const int tk0 = tid >> 4;       // WT=1 map: 0..7 k start

    v8f acc[2][2];
    #pragma unroll
    for (int mi = 0; mi < 2; ++mi)
        #pragma unroll
        for (int ni = 0; ni < 2; ++ni)
            acc[mi][ni] = v8f{0.f,0.f,0.f,0.f,0.f,0.f,0.f,0.f};

    for (int k0 = 0; k0 < K; k0 += TILE_K) {
        // ---- stage A tile 64x32 -> As[m][k] (M, K always full) ----
        #pragma unroll
        for (int i = 0; i < 4; ++i) {
            const int row = sr + 16 * i;
            const float4 v = *(const float4*)
                (A + (size_t)(mBase + row) * lda + k0 + sc * 4);
            *(uint2*)&As[row][sc * 4] =
                uint2{pack_bf16x2(v.x, v.y), pack_bf16x2(v.z, v.w)};
        }

        // ---- stage W tile -> BsT[n][k] ----
        if constexpr (WT == 0) {   // W[N,K]: coalesced on K
            #pragma unroll
            for (int i = 0; i < 4; ++i) {
                const int n  = sr + 16 * i;
                const int gn = nBase + n;
                if (FULL_N || gn < N) {
                    const float4 v = *(const float4*)
                        (W + (size_t)gn * ldw + k0 + sc * 4);
                    *(uint2*)&BsT[n][sc * 4] =
                        uint2{pack_bf16x2(v.x, v.y), pack_bf16x2(v.z, v.w)};
                } else {
                    *(uint2*)&BsT[n][sc * 4] = uint2{0u, 0u};
                }
            }
        } else {                   // W[K,N]: coalesced on N, transpose-scatter
            #pragma unroll
            for (int i = 0; i < 4; ++i) {
                const int kk = tk0 + 8 * i;
                if (FULL_N || nBase + tn4 * 4 + 3 < N) {
                    const float4 v = *(const float4*)
                        (W + (size_t)(k0 + kk) * ldw + nBase + tn4 * 4);
                    BsT[tn4 * 4 + 0][kk] = f2bf(v.x);
                    BsT[tn4 * 4 + 1][kk] = f2bf(v.y);
                    BsT[tn4 * 4 + 2][kk] = f2bf(v.z);
                    BsT[tn4 * 4 + 3][kk] = f2bf(v.w);
                } else {
                    #pragma unroll
                    for (int j = 0; j < 4; ++j) {
                        const int gn = nBase + tn4 * 4 + j;
                        BsT[tn4 * 4 + j][kk] =
                            (gn < N) ? f2bf(W[(size_t)(k0 + kk) * ldw + gn])
                                     : f2bf(0.0f);
                    }
                }
            }
        }

        // prefetch next K-tile while this one is consumed
        if (k0 + TILE_K < K) {
            __builtin_prefetch(A + (size_t)(mBase + sr) * lda + k0 + TILE_K + sc * 4, 0, 3);
            if constexpr (WT == 0)
                __builtin_prefetch(W + (size_t)(nBase + sr) * ldw + k0 + TILE_K + sc * 4, 0, 3);
            else
                __builtin_prefetch(W + (size_t)(k0 + TILE_K + tk0) * ldw + nBase + tn4 * 4, 0, 3);
        }

        __syncthreads();

        // ---- per-lane fragments: two contiguous 16B chunks each ----
        v16bf afrag[2], bfrag[2];
        #pragma unroll
        for (int mi = 0; mi < 2; ++mi) {
            const int row = wm + mi * 16 + l16;
            const v8bf lo = *(const v8bf*)&As[row][kb];
            const v8bf hi = *(const v8bf*)&As[row][kb + 16];
            afrag[mi] = __builtin_shufflevector(lo, hi,
                0,1,2,3,4,5,6,7,8,9,10,11,12,13,14,15);
        }
        #pragma unroll
        for (int ni = 0; ni < 2; ++ni) {
            const int col = wn + ni * 16 + l16;
            const v8bf lo = *(const v8bf*)&BsT[col][kb];
            const v8bf hi = *(const v8bf*)&BsT[col][kb + 16];
            bfrag[ni] = __builtin_shufflevector(lo, hi,
                0,1,2,3,4,5,6,7,8,9,10,11,12,13,14,15);
        }

        #pragma unroll
        for (int mi = 0; mi < 2; ++mi)
            #pragma unroll
            for (int ni = 0; ni < 2; ++ni)
                acc[mi][ni] = __builtin_amdgcn_wmma_f32_16x16x32_bf16(
                    false, afrag[mi], false, bfrag[ni],
                    (short)0, acc[mi][ni], false, false);

        __syncthreads();
    }

    // ---- epilogue: C/D layout lane->N, VGPR r -> M = halfSel*8 + r ----
    #pragma unroll
    for (int mi = 0; mi < 2; ++mi) {
        #pragma unroll
        for (int ni = 0; ni < 2; ++ni) {
            const int gn = nBase + wn + ni * 16 + l16;
            if (!FULL_N && gn >= N) continue;
            const float bv = BIAS ? bias[gn] : 0.0f;
            #pragma unroll
            for (int r = 0; r < 8; ++r) {
                const int gm = mBase + wm + mi * 16 + halfSel * 8 + r;
                float v = acc[mi][ni][r];
                if constexpr (BIAS)     v += bv;
                if constexpr (ACT == 1) v = silu_f(v);
                if constexpr (ACT == 2) v = softplus_f(v);
                if constexpr (RES)      v += residual[(size_t)gm * ldc + gn];
                C[(size_t)gm * ldc + gn] = v;
            }
        }
    }
}

// ---------------------------------------------------------------------------
// RMSNorm. One block per output row; [b, tOff + t] row remap serves both the
// per-layer norm (Tout==Tin, tOff=0) and the final norm (Tout=S, tOff=P).
// ---------------------------------------------------------------------------
__global__ __launch_bounds__(256) void rmsnorm_kernel(
    const float* __restrict__ in, const float* __restrict__ w,
    float* __restrict__ out, int D, int Tin, int Tout, int tOff)
{
    __shared__ float red[256];
    const int rowOut = blockIdx.x;
    const int b = rowOut / Tout;
    const int t = rowOut % Tout;
    const int rowIn = b * Tin + tOff + t;

    const float* x = in + (size_t)rowIn * D;
    float s = 0.0f;
    for (int i = threadIdx.x; i < D; i += 256) { float v = x[i]; s += v * v; }
    red[threadIdx.x] = s;
    __syncthreads();
    for (int off = 128; off > 0; off >>= 1) {
        if (threadIdx.x < off) red[threadIdx.x] += red[threadIdx.x + off];
        __syncthreads();
    }
    const float inv = rsqrtf(red[0] / (float)D + 1e-5f);
    float* o = out + (size_t)rowOut * D;
    for (int i = threadIdx.x; i < D; i += 256) o[i] = x[i] * inv * w[i];
}

// h[b, t, :] = t < P ? p[b, t, :] : x[b, t-P, :]
__global__ void concat_kernel(const float* __restrict__ p,
                              const float* __restrict__ x,
                              float* __restrict__ h,
                              int B, int P, int S, int D)
{
    const int T = P + S;
    const size_t total = (size_t)B * T * D;
    size_t idx = (size_t)blockIdx.x * blockDim.x + threadIdx.x;
    if (idx >= total) return;
    const int d = idx % D;
    const int t = (idx / D) % T;
    const int b = idx / ((size_t)D * T);
    h[idx] = (t < P) ? p[((size_t)b * P + t) * D + d]
                     : x[((size_t)b * S + (t - P)) * D + d];
}

// Depthwise causal conv1d (DC=4) + bias + SiLU over the xi half of xz.
__global__ void conv_silu_kernel(const float* __restrict__ xz,
                                 const float* __restrict__ cw,
                                 const float* __restrict__ cb,
                                 float* __restrict__ xc,
                                 int B, int T, int DI, int twoDI)
{
    const size_t total = (size_t)B * T * DI;
    size_t idx = (size_t)blockIdx.x * blockDim.x + threadIdx.x;
    if (idx >= total) return;
    const int e  = idx % DI;
    const int bt = idx / DI;
    const int t  = bt % T;
    const int b  = bt / T;
    float acc = cb[e];
    #pragma unroll
    for (int k = 0; k < 4; ++k) {
        const int tt = t - 3 + k;
        if (tt >= 0)
            acc += cw[e * 4 + k] * xz[((size_t)b * T + tt) * twoDI + e];
    }
    xc[idx] = silu_f(acc);
}

// Fused selective scan. One lane per (b, channel e): 16 SSM states live in
// registers; exp(dt*A) computed inline; y gated by SiLU(z) and D skip.
__global__ __launch_bounds__(256) void scan_kernel(
    const float* __restrict__ xz,    // [B,T,2*DI] (z at +DI)
    const float* __restrict__ xc,    // [B,T,DI]   post-conv SiLU
    const float* __restrict__ xdbl,  // [B,T,96]   (B at +64, C at +80)
    const float* __restrict__ dtb,   // [B,T,DI]   softplus dt
    const float* __restrict__ A_log, // [DI,16]
    const float* __restrict__ Dp,    // [DI]
    float* __restrict__ yb,          // [B,T,DI]
    int B, int T, int DI)
{
    const int idx = blockIdx.x * 256 + threadIdx.x;
    if (idx >= B * DI) return;
    const int e = idx % DI;
    const int b = idx / DI;

    float Arow[16];
    #pragma unroll
    for (int s = 0; s < 16; ++s) Arow[s] = -__expf(A_log[(size_t)e * 16 + s]);
    const float Dv = Dp[e];

    float h[16];
    #pragma unroll
    for (int s = 0; s < 16; ++s) h[s] = 0.0f;

    for (int t = 0; t < T; ++t) {
        const size_t bt = (size_t)b * T + t;
        const float dtv = dtb[bt * DI + e];
        const float xiv = xc[bt * DI + e];
        const float* Bm = xdbl + bt * 96 + 64;
        const float* Cm = xdbl + bt * 96 + 80;
        float y = 0.0f;
        #pragma unroll
        for (int s = 0; s < 16; ++s) {
            h[s] = __expf(dtv * Arow[s]) * h[s] + dtv * Bm[s] * xiv;
            y += h[s] * Cm[s];
        }
        const float zv = xz[bt * (size_t)(2 * DI) + DI + e];
        yb[bt * DI + e] = (y + xiv * Dv) * silu_f(zv);
    }
}

// ---------------------------------------------------------------------------
// Host side
// ---------------------------------------------------------------------------
template <bool FULL_N, int WT, int ACT, bool BIAS, bool RES>
static inline void launch_gemm(const float* A, int lda,
                               const float* W, int ldw,
                               float* C, int ldc,
                               const float* bias, const float* res,
                               int M, int N, int K,
                               hipStream_t stream)
{
    dim3 grid((N + TILE_N - 1) / TILE_N, (M + TILE_M - 1) / TILE_M);
    gemm_wmma_kernel<FULL_N, WT, ACT, BIAS, RES>
        <<<grid, 128, 0, stream>>>(A, lda, W, ldw, C, ldc, bias, res, N, K);
}

extern "C" void kernel_launch(void* const* d_in, const int* in_sizes, int n_in,
                              void* d_out, int out_size, void* d_ws, size_t ws_size,
                              hipStream_t stream)
{
    (void)in_sizes; (void)n_in; (void)out_size; (void)ws_size;

    constexpr int B = 2, S = 512, P = 64, D = 1024, L = 4;
    constexpr int DI = 2048, DS = 16, DC = 4, DR = 64, PCD = 768;
    constexpr int T = P + S;                 // 576
    constexpr int M = B * T;                 // 1152 rows (multiple of 64)
    constexpr int XPN = DR + 2 * DS;         // 96

    const float* x        = (const float*)d_in[0];
    const float* pc       = (const float*)d_in[1];
    const float* pre_w1   = (const float*)d_in[2];
    const float* pre_w2   = (const float*)d_in[3];
    const float* norm_w   = (const float*)d_in[4];
    const float* in_proj  = (const float*)d_in[5];
    const float* conv_w   = (const float*)d_in[6];
    const float* conv_b   = (const float*)d_in[7];
    const float* x_proj   = (const float*)d_in[8];
    const float* dt_projw = (const float*)d_in[9];
    const float* dt_projb = (const float*)d_in[10];
    const float* A_log    = (const float*)d_in[11];
    const float* D_param  = (const float*)d_in[12];
    const float* out_proj = (const float*)d_in[13];
    const float* fnorm_w  = (const float*)d_in[14];
    float* out = (float*)d_out;

    // workspace carve-up (floats)
    float* ws   = (float*)d_ws;
    float* h    = ws;                         // [M, D]
    float* xn   = h    + (size_t)M * D;       // [M, D]
    float* xzb  = xn   + (size_t)M * D;       // [M, 2*DI]
    float* xcb  = xzb  + (size_t)M * 2 * DI;  // [M, DI]
    float* xdb  = xcb  + (size_t)M * DI;      // [M, 96]
    float* dtb  = xdb  + (size_t)M * XPN;     // [M, DI]
    float* ybf  = dtb  + (size_t)M * DI;      // [M, DI]
    float* t0   = ybf  + (size_t)M * DI;      // [B*P, D]
    float* pbuf = t0   + (size_t)B * P * D;   // [B*P, D]

    // ---- prepend MLP: silu(pc @ W1) @ W2 ----
    launch_gemm<true, 1, 1, false, false>(pc, PCD, pre_w1, D, t0, D,
                                          nullptr, nullptr, B * P, D, PCD, stream);
    launch_gemm<true, 1, 0, false, false>(t0, D, pre_w2, D, pbuf, D,
                                          nullptr, nullptr, B * P, D, D, stream);

    // ---- h = concat(p, x) along time ----
    {
        size_t total = (size_t)B * T * D;
        concat_kernel<<<(total + 255) / 256, 256, 0, stream>>>(
            pbuf, x, h, B, P, S, D);
    }

    // ---- layers ----
    for (int l = 0; l < L; ++l) {
        // rmsnorm(h) -> xn
        rmsnorm_kernel<<<M, 256, 0, stream>>>(h, norm_w + (size_t)l * D,
                                              xn, D, T, T, 0);
        // xz = xn @ in_proj^T      [M, 2*DI]
        launch_gemm<true, 0, 0, false, false>(
            xn, D, in_proj + (size_t)l * 2 * DI * D, D,
            xzb, 2 * DI, nullptr, nullptr, M, 2 * DI, D, stream);
        // causal depthwise conv + SiLU on xi half -> xcb
        {
            size_t total = (size_t)M * DI;
            conv_silu_kernel<<<(total + 255) / 256, 256, 0, stream>>>(
                xzb, conv_w + (size_t)l * DI * DC, conv_b + (size_t)l * DI,
                xcb, B, T, DI, 2 * DI);
        }
        // xdbl = xc @ x_proj^T     [M, 96]  (only GEMM with partial N)
        launch_gemm<false, 0, 0, false, false>(
            xcb, DI, x_proj + (size_t)l * XPN * DI, DI,
            xdb, XPN, nullptr, nullptr, M, XPN, DI, stream);
        // dt = softplus(xdbl[:, :64] @ dt_proj^T + b)   [M, DI]
        launch_gemm<true, 0, 2, true, false>(
            xdb, XPN, dt_projw + (size_t)l * DI * DR, DR,
            dtb, DI, dt_projb + (size_t)l * DI, nullptr, M, DI, DR, stream);
        // fused selective scan -> ybf (includes D-skip and SiLU(z) gate)
        scan_kernel<<<(B * DI + 255) / 256, 256, 0, stream>>>(
            xzb, xcb, xdb, dtb,
            A_log + (size_t)l * DI * DS, D_param + (size_t)l * DI,
            ybf, B, T, DI);
        // h = h + ybf @ out_proj^T
        launch_gemm<true, 0, 0, false, true>(
            ybf, DI, out_proj + (size_t)l * D * DI, DI,
            h, D, nullptr, /*residual=*/h, M, D, DI, stream);
    }

    // ---- final rmsnorm, drop the P prepended rows ----
    rmsnorm_kernel<<<B * S, 256, 0, stream>>>(h, fnorm_w, out, D, T, S, P);
}
